// SVTRAttention_3951369912780
// MI455X (gfx1250) — matrix operand order repaired and checked by reading, not compiled, not run on hardware
//
#include <hip/hip_runtime.h>
#include <hip/hip_bf16.h>

// ---------------- WMMA types & helpers (gfx1250, wave32) ----------------
typedef __attribute__((ext_vector_type(16))) __bf16 v16bf;
typedef __attribute__((ext_vector_type(8)))  float  v8f;
typedef __attribute__((ext_vector_type(2)))  __bf16 bf16x2;
typedef unsigned int u32x4 __attribute__((ext_vector_type(4)));
typedef int          i32x4 __attribute__((ext_vector_type(4)));
typedef int          i32x8 __attribute__((ext_vector_type(8)));

#define SCALE_Q 0.17677669529663687f   // 32^-0.5
#define VT_LD   520                    // 512 + 8 pad (bank spread)

__device__ __forceinline__ v8f wmma_bf16(v16bf a, v16bf b, v8f c) {
  return __builtin_amdgcn_wmma_f32_16x16x32_bf16(false, a, false, b, (short)0, c,
                                                 false, false);
}

__device__ __forceinline__ int iabs_(int x) { return x < 0 ? -x : x; }

// ---- Tensor Data Mover: 1D bulk copy global -> LDS (2-byte elements) ----
// D# per CDNA5 ISA: group0 {count=1, lds_addr, global_addr,
// type=2}; group1 {data_size=2B, tensor_dim0=tile_dim0=nelem, dim1=1, stride=nelem}.
__device__ __forceinline__ void tdm_copy_b16_to_lds(const void* gptr,
                                                    unsigned lds_byte_off,
                                                    unsigned nelem /* <= 65535 */) {
  unsigned long long ga = (unsigned long long)(uintptr_t)gptr;
  u32x4 g0;
  g0[0] = 1u;                                              // count=1 (valid user D#)
  g0[1] = lds_byte_off;                                    // lds_addr
  g0[2] = (unsigned)(ga & 0xFFFFFFFFu);                    // global_addr[31:0]
  g0[3] = (unsigned)((ga >> 32) & 0x01FFFFFFu) | (2u << 30); // global_addr[56:32], type=2
  i32x8 g1;
  g1[0] = (1 << 16);                                       // wg_mask=0, data_size=1 (2B)
  g1[1] = (int)((nelem & 0xFFFFu) << 16);                  // tensor_dim0[15:0]
  g1[2] = (int)((nelem >> 16) & 0xFFFFu) | (1 << 16);      // tensor_dim0[31:16], tensor_dim1=1
  g1[3] = (int)((nelem & 0xFFFFu) << 16);                  // tensor_dim1 hi=0, tile_dim0=nelem
  g1[4] = 1;                                               // tile_dim1=1, tile_dim2=0
  g1[5] = (int)nelem;                                      // tensor_dim0_stride lo32
  g1[6] = 0;                                               // stride hi / dim1_stride lo
  g1[7] = 0;
  i32x4 z4 = {0, 0, 0, 0};
#if defined(__clang_major__) && (__clang_major__ >= 23)
  i32x8 z8 = {0, 0, 0, 0, 0, 0, 0, 0};
  __builtin_amdgcn_tensor_load_to_lds(g0, g1, z4, z4, z8, 0);
#else
  __builtin_amdgcn_tensor_load_to_lds(g0, g1, z4, z4, 0);
#endif
}

// A fragment (16x32 bf16): lane holds row m=lane&15.
// K contiguous: lanes 0-15 -> K=0..15, lanes 16-31 -> K=16..31
__device__ __forceinline__ v16bf load_a_from_f32(const float* __restrict__ src,
                                                 int ldk, int row, int kk, int lane) {
  v16bf a;
  const float* p = src + (size_t)row * ldk + kk;
  const int base = (lane < 16) ? 0 : 16;
#pragma unroll
  for (int g = 0; g < 8; ++g) {
    int K = base + 2 * g;
    float2 f = *(const float2*)(p + K);
    a[2 * g]     = (__bf16)f.x;
    a[2 * g + 1] = (__bf16)f.y;
  }
  return a;
}

__device__ __forceinline__ v16bf load_a_from_bf16(const __bf16* __restrict__ src,
                                                  int ldk, int row, int kk, int lane) {
  v16bf a;
  const __bf16* p = src + (size_t)row * ldk + kk;
  const int base = (lane < 16) ? 0 : 16;
#pragma unroll
  for (int g = 0; g < 8; ++g) {
    int K = base + 2 * g;
    bf16x2 f = *(const bf16x2*)(p + K);
    a[2 * g]     = f.x;
    a[2 * g + 1] = f.y;
  }
  return a;
}

// A fragment sourced from an f32 LDS strip (P matrix after softmax)
__device__ __forceinline__ v16bf load_a_from_lds_f32(const float* s, int ldk,
                                                     int row, int kk, int lane) {
  v16bf a;
  const float* p = s + (size_t)row * ldk + kk;
  const int base = (lane < 16) ? 0 : 16;
#pragma unroll
  for (int g = 0; g < 8; ++g) {
    int K = base + 2 * g;
    float2 f = *(const float2*)(p + K);
    a[2 * g]     = (__bf16)f.x;
    a[2 * g + 1] = (__bf16)f.y;
  }
  return a;
}

// B fragment (32x16 bf16), B[k][n] = W[n][k] (row-major W, transpose-on-load).
// lane holds column n=lane&15; K contiguous: lanes 0-15 -> K=0..15, lanes 16-31 -> 16..31
__device__ __forceinline__ v16bf load_bT_from_f32(const float* __restrict__ w,
                                                  int ldk, int n, int kk, int lane) {
  v16bf b;
  const float* p = w + (size_t)n * ldk + kk + ((lane < 16) ? 0 : 16);
#pragma unroll
  for (int g = 0; g < 8; ++g) {
    float2 f = *(const float2*)(p + 2 * g);
    b[2 * g]     = (__bf16)f.x;
    b[2 * g + 1] = (__bf16)f.y;
  }
  return b;
}

__device__ __forceinline__ v16bf load_bT_from_bf16(const __bf16* m, int ldk, int n,
                                                   int kk, int lane) {
  v16bf b;
  const __bf16* p = m + (size_t)n * ldk + kk + ((lane < 16) ? 0 : 16);
#pragma unroll
  for (int g = 0; g < 8; ++g) {
    bf16x2 f = *(const bf16x2*)(p + 2 * g);
    b[2 * g]     = f.x;
    b[2 * g + 1] = f.y;
  }
  return b;
}

// B fragment for PV from LDS V^T [32][VT_LD]: B[k][n] = Vt[n][k], contiguous in K
__device__ __forceinline__ v16bf load_b_from_vt(const __bf16* vt, int n0, int kk,
                                                int lane) {
  v16bf b;
  const __bf16* p = vt + (size_t)(n0 + (lane & 15)) * VT_LD + kk +
                    ((lane < 16) ? 0 : 16);
#pragma unroll
  for (int g = 0; g < 8; ++g) {
    bf16x2 f = *(const bf16x2*)(p + 2 * g);
    b[2 * g]     = f.x;
    b[2 * g + 1] = f.y;
  }
  return b;
}

// ---------------- Kernel 1: QKV projection GEMM ----------------
__global__ __launch_bounds__(256) void svtr_qkv_gemm(
    const float* __restrict__ x, const float* __restrict__ wqkv,
    __bf16* __restrict__ qw, __bf16* __restrict__ kw, __bf16* __restrict__ vw) {
  const int lane = threadIdx.x & 31;
  const int wave = threadIdx.x >> 5;
  const int tile = blockIdx.x * 8 + wave;   // 98304 tiles total
  const int tm = tile / 48, tn = tile - tm * 48;
  const int m0 = tm * 16, n0 = tn * 16;

  v8f acc = {};
#pragma unroll
  for (int kk = 0; kk < 256; kk += 32) {
    v16bf a = load_a_from_f32(x, 256, m0 + (lane & 15), kk, lane);
    v16bf b = load_bT_from_f32(wqkv, 256, n0 + (lane & 15), kk, lane);
    acc = wmma_bf16(a, b, acc);
  }

  const int j = n0 + (lane & 15);      // [0,768)
  const int part = j >> 8;             // 0=q 1=k 2=v
  const int head = (j & 255) >> 5;
  const int d = j & 31;
  __bf16* dst = (part == 0) ? qw : (part == 1) ? kw : vw;
  const float mul = (part == 0) ? SCALE_Q : 1.0f;
#pragma unroll
  for (int r = 0; r < 8; ++r) {
    int m = m0 + r + ((lane < 16) ? 0 : 8);   // [0, 32768)
    int bidx = m >> 9;
    int tok = m & 511;
    dst[(((size_t)bidx * 8 + head) * 512 + tok) * 32 + d] = (__bf16)(acc[r] * mul);
  }
}

// ---------------- Kernel 2: local-window attention per (b,h) ----------------
// LDS: 4x 16x512 f32 S strips (128KB) + K strip via TDM (32KB) + V^T (32.5KB)
__global__ __launch_bounds__(128) void svtr_attention(
    const __bf16* __restrict__ qw, const __bf16* __restrict__ kw,
    const __bf16* __restrict__ vw, __bf16* __restrict__ ow) {
  __shared__ float  Sb[4][16 * 512];
  __shared__ __bf16 Ks[512 * 32];
  __shared__ __bf16 Vt[32 * VT_LD];

  const int lane = threadIdx.x & 31;
  const int wave = threadIdx.x >> 5;
  const int bh = blockIdx.x;              // 512 = B*H
  const int b = bh >> 3, h = bh & 7;
  const size_t base = (size_t)bh * 512 * 32;
  float* S = Sb[wave];

  // ---- stage K strip via Tensor Data Mover (one wave issues; TENSORcnt wait) ----
  if (wave == 0) {
    unsigned lds_off = (unsigned)(uintptr_t)(void*)&Ks[0];
    tdm_copy_b16_to_lds(kw + base, lds_off, 512 * 32);
    __builtin_amdgcn_s_wait_tensorcnt(0);
  }
  // ---- stage V transposed: Vt[d][key] (coalesced global reads) ----
  for (int i = 0; i < 128; ++i) {
    int e = i * 128 + threadIdx.x;        // [0, 16384)
    int key = e >> 5, d = e & 31;
    Vt[d * VT_LD + key] = vw[base + e];
  }
  __syncthreads();

  for (int qb = wave; qb < 32; qb += 4) {
    const int m0 = qb * 16;
    // q A-fragment: full head_dim (K=32) in one fragment, reused for all key tiles
    v16bf qa = load_a_from_bf16(qw + base, 32, m0 + (lane & 15), 0, lane);

    // ---- S = (q*scale) k^T, masked, into LDS ----
    for (int kt = 0; kt < 32; ++kt) {
      v16bf kb = load_bT_from_bf16(Ks, 32, kt * 16 + (lane & 15), 0, lane);
      v8f z = {};
      v8f s = wmma_bf16(qa, kb, z);
      const int key = kt * 16 + (lane & 15);
      const int kh = key >> 6, kwid = key & 63;
#pragma unroll
      for (int r = 0; r < 8; ++r) {
        int row = r + ((lane < 16) ? 0 : 8);
        int qpos = m0 + row;
        int qh = qpos >> 6, qwid = qpos & 63;
        bool ok = (iabs_(qh - kh) <= 3) && (iabs_(qwid - kwid) <= 5);
        S[row * 512 + key] = ok ? s[r] : -3.0e38f;
      }
    }
    asm volatile("s_wait_dscnt 0" ::: "memory");  // intra-wave LDS visibility

    // ---- softmax over 16 rows; lane pair (l, l+16) splits the 512 columns ----
    {
      const int row = lane & 15;
      float* Sr = S + row * 512 + ((lane >> 4) << 8);   // 256 cols per lane
      float mx = -3.0e38f;
      for (int c = 0; c < 256; ++c) mx = fmaxf(mx, Sr[c]);
      mx = fmaxf(mx, __shfl_xor(mx, 16, 32));
      float sum = 0.0f;
      for (int c = 0; c < 256; ++c) { float e = __expf(Sr[c] - mx); Sr[c] = e; sum += e; }
      sum += __shfl_xor(sum, 16, 32);
      const float inv = 1.0f / sum;
      for (int c = 0; c < 256; ++c) Sr[c] *= inv;
    }
    asm volatile("s_wait_dscnt 0" ::: "memory");

    // ---- O = P @ V : M=16, N=32 (2 tiles), K=512 (16 steps); V^T from LDS ----
    v8f acc0 = {}, acc1 = {};
    for (int ks = 0; ks < 16; ++ks) {
      v16bf pa = load_a_from_lds_f32(S, 512, lane & 15, ks * 32, lane);
      v16bf b0 = load_b_from_vt(Vt, 0, ks * 32, lane);
      v16bf b1 = load_b_from_vt(Vt, 16, ks * 32, lane);
      acc0 = wmma_bf16(pa, b0, acc0);
      acc1 = wmma_bf16(pa, b1, acc1);
    }
    // store to attn-out workspace, layout [B, tok, head*32+d] bf16
#pragma unroll
    for (int r = 0; r < 8; ++r) {
      int row = r + ((lane < 16) ? 0 : 8);
      int tok = m0 + row;
      size_t rb = ((size_t)b * 512 + tok) * 256 + h * 32;
      ow[rb + (lane & 15)]      = (__bf16)acc0[r];
      ow[rb + 16 + (lane & 15)] = (__bf16)acc1[r];
    }
  }
}

// ---------------- Kernel 3: output projection + bias ----------------
__global__ __launch_bounds__(256) void svtr_proj_gemm(
    const __bf16* __restrict__ ao, const float* __restrict__ wproj,
    const float* __restrict__ bias, float* __restrict__ out) {
  const int lane = threadIdx.x & 31;
  const int wave = threadIdx.x >> 5;
  const int tile = blockIdx.x * 8 + wave;   // 32768 tiles
  const int tm = tile >> 4, tn = tile & 15;
  const int m0 = tm * 16, n0 = tn * 16;

  v8f acc = {};
#pragma unroll
  for (int kk = 0; kk < 256; kk += 32) {
    v16bf a = load_a_from_bf16(ao, 256, m0 + (lane & 15), kk, lane);
    v16bf b = load_bT_from_f32(wproj, 256, n0 + (lane & 15), kk, lane);
    acc = wmma_bf16(a, b, acc);
  }
  const int n = n0 + (lane & 15);
  const float bv = bias[n];
#pragma unroll
  for (int r = 0; r < 8; ++r) {
    int m = m0 + r + ((lane < 16) ? 0 : 8);
    out[(size_t)m * 256 + n] = acc[r] + bv;
  }
}

// ---------------- launch ----------------
extern "C" void kernel_launch(void* const* d_in, const int* in_sizes, int n_in,
                              void* d_out, int out_size, void* d_ws, size_t ws_size,
                              hipStream_t stream) {
  (void)in_sizes; (void)n_in; (void)out_size; (void)ws_size;
  const float* x     = (const float*)d_in[0];   // [64,512,256]
  const float* wqkv  = (const float*)d_in[1];   // [768,256]
  const float* wproj = (const float*)d_in[2];   // [256,256]
  const float* bproj = (const float*)d_in[3];   // [256]
  float* out = (float*)d_out;                   // [64,512,256] f32

  // workspace layout (bf16): q | k | v | attn_out ; each 8,388,608 elems (16 MiB)
  __bf16* q  = (__bf16*)d_ws;
  __bf16* k  = q + (size_t)8388608;
  __bf16* v  = k + (size_t)8388608;
  __bf16* ao = v + (size_t)8388608;

  svtr_qkv_gemm<<<12288, 256, 0, stream>>>(x, wqkv, q, k, v);
  svtr_attention<<<512, 128, 0, stream>>>(q, k, v, ao);
  svtr_proj_gemm<<<4096, 256, 0, stream>>>(ao, wproj, bproj, out);
}
